// CondLaneHead_26946624815738
// MI455X (gfx1250) — matrix operand order repaired and checked by reading, not compiled, not run on hardware
//
#include <hip/hip_runtime.h>

typedef __attribute__((ext_vector_type(16))) _Float16 v16h;
typedef __attribute__((ext_vector_type(8)))  float    v8f;
typedef __attribute__((ext_vector_type(4)))  float    v4f;

#define HH 160
#define WW 256
#define HWSZ (HH*WW)
#define FSTR 104   // feat LDS row stride in halves (96 padded-CIN + 8 pad)
#define HSTR 72    // hidden LDS row stride in halves (64 + 8 pad)
#define NPARAMS 8513

union Frag  { v16h h; v4f f4[2]; };

__global__ __launch_bounds__(256) void condlane_wmma_kernel(
    const float* __restrict__ x,        // [4, 64, 160, 256]
    const float* __restrict__ params,   // [32, 8513]
    float* __restrict__ out)            // [1, 32, 160, 256]
{
    __shared__ _Float16 s_feat[256 * FSTR];   // features (K=96 padded); later re-used as hdn1
    __shared__ _Float16 s_hdn0[256 * HSTR];
    __shared__ _Float16 s_w0[64 * FSTR];
    __shared__ _Float16 s_w1[64 * HSTR];
    __shared__ float s_b0[64], s_b1[64], s_w2[64];
    __shared__ float s_b2;

    const int row  = blockIdx.x;   // 0..159 (y); one 256-pixel tile == one image row
    const int inst = blockIdx.y;   // 0..31
    const int img  = inst >> 3;    // 8 instances per image
    const int tid  = threadIdx.x;
    const float* P = params + (size_t)inst * NPARAMS;

    // ---- stage weights to LDS (f16, zero-padded K) ----
    for (int idx = tid; idx < 64 * FSTR; idx += 256) {
        int r = idx / FSTR, c = idx - r * FSTR;
        float v = (c < 66) ? P[r * 66 + c] : 0.0f;        // W0: [64][66]
        s_w0[idx] = (_Float16)v;
    }
    for (int idx = tid; idx < 64 * HSTR; idx += 256) {
        int r = idx / HSTR, c = idx - r * HSTR;
        float v = (c < 64) ? P[4224 + r * 64 + c] : 0.0f; // W1: [64][64]
        s_w1[idx] = (_Float16)v;
    }
    if (tid < 64) {
        s_w2[tid] = P[8320 + tid];
        s_b0[tid] = P[8384 + tid];
        s_b1[tid] = P[8448 + tid];
        if (tid == 0) s_b2 = P[8512] - 2.19f;             // is_mask bias shift
    }

    // ---- stage feature tile to LDS: pixel-major [256][FSTR], K order = {locx, locy, x[0..63], 0...}
    {
        _Float16* f = s_feat + tid * FSTR;
        f[0] = (_Float16)(float)tid;   // loc_x (unnormalized, stride 1)
        f[1] = (_Float16)(float)row;   // loc_y
        const float* xb = x + (size_t)img * (64 * HWSZ) + row * WW + tid;
        #pragma unroll
        for (int c = 0; c < 64; ++c)
            f[2 + c] = (_Float16)xb[(size_t)c * HWSZ];    // coalesced across lanes
        #pragma unroll
        for (int c = 66; c < FSTR; ++c) f[c] = (_Float16)0.0f;
    }
    __syncthreads();

    const int lane  = tid & 31;
    const int wave  = tid >> 5;
    const int mt    = wave & 3;        // M tile: output chans [16*mt, 16*mt+16)
    const int nh    = wave >> 2;       // N half: N-tiles nh*8 .. nh*8+7
    const int ln    = lane & 15;
    const int hi    = lane >> 4;
    const int mrow  = mt * 16 + ln;    // A-fragment row this lane supplies
    const int mbase = mt * 16 + hi * 8;// C/D row base this lane holds

    // ================= layer 0: hdn0 = relu(W0 * feat + b0), K = 96 =================
    Frag a0[3];
    #pragma unroll
    for (int ks = 0; ks < 3; ++ks) {
        const _Float16* pa = s_w0 + mrow * FSTR + ks * 32 + hi * 16;
        a0[ks].f4[0] = *(const v4f*)pa;          // K = ks*32 + hi*16 + 0..7
        a0[ks].f4[1] = *(const v4f*)(pa + 8);    // K = ks*32 + hi*16 + 8..15
    }
    #pragma unroll
    for (int nt = 0; nt < 8; ++nt) {
        const int pix = (nh * 8 + nt) * 16 + ln;
        v8f acc = {0.f,0.f,0.f,0.f,0.f,0.f,0.f,0.f};
        #pragma unroll
        for (int ks = 0; ks < 3; ++ks) {
            Frag b;
            const _Float16* pb = s_feat + pix * FSTR + ks * 32 + hi * 16;
            b.f4[0] = *(const v4f*)pb;           // K = ks*32 + hi*16 + 0..7
            b.f4[1] = *(const v4f*)(pb + 8);     // K = ks*32 + hi*16 + 8..15
            acc = __builtin_amdgcn_wmma_f32_16x16x32_f16(
                false, a0[ks].h, false, b.h, (short)0, acc, false, false);
        }
        #pragma unroll
        for (int r = 0; r < 8; r += 2) {
            float v0 = fmaxf(acc[r]     + s_b0[mbase + r],     0.0f);
            float v1 = fmaxf(acc[r + 1] + s_b0[mbase + r + 1], 0.0f);
            union { _Float16 h[2]; unsigned u; } pk;
            pk.h[0] = (_Float16)v0; pk.h[1] = (_Float16)v1;
            *(unsigned*)(s_hdn0 + pix * HSTR + mbase + r) = pk.u;
        }
    }
    __syncthreads();

    // ================= layer 1: hdn1 = relu(W1 * hdn0 + b1), K = 64 =================
    Frag a1[2];
    #pragma unroll
    for (int ks = 0; ks < 2; ++ks) {
        const _Float16* pa = s_w1 + mrow * HSTR + ks * 32 + hi * 16;
        a1[ks].f4[0] = *(const v4f*)pa;
        a1[ks].f4[1] = *(const v4f*)(pa + 8);
    }
    #pragma unroll
    for (int nt = 0; nt < 8; ++nt) {
        const int pix = (nh * 8 + nt) * 16 + ln;
        v8f acc = {0.f,0.f,0.f,0.f,0.f,0.f,0.f,0.f};
        #pragma unroll
        for (int ks = 0; ks < 2; ++ks) {
            Frag b;
            const _Float16* pb = s_hdn0 + pix * HSTR + ks * 32 + hi * 16;
            b.f4[0] = *(const v4f*)pb;
            b.f4[1] = *(const v4f*)(pb + 8);
            acc = __builtin_amdgcn_wmma_f32_16x16x32_f16(
                false, a1[ks].h, false, b.h, (short)0, acc, false, false);
        }
        #pragma unroll
        for (int r = 0; r < 8; r += 2) {
            float v0 = fmaxf(acc[r]     + s_b1[mbase + r],     0.0f);
            float v1 = fmaxf(acc[r + 1] + s_b1[mbase + r + 1], 0.0f);
            union { _Float16 h[2]; unsigned u; } pk;
            pk.h[0] = (_Float16)v0; pk.h[1] = (_Float16)v1;
            // hdn1 aliases s_feat (safe: all feat reads finished before prior barrier)
            *(unsigned*)(s_feat + pix * HSTR + mbase + r) = pk.u;
        }
    }
    __syncthreads();

    // ================= layer 2: out = w2 . hdn1 + (b2 - 2.19) =================
    {
        const _Float16* hp = s_feat + tid * HSTR;
        float acc = 0.0f;
        #pragma unroll
        for (int c = 0; c < 64; c += 8) {
            union { v4f f; _Float16 h[8]; } u;
            u.f = *(const v4f*)(hp + c);
            #pragma unroll
            for (int j = 0; j < 8; ++j)
                acc = fmaf(s_w2[c + j], (float)u.h[j], acc);
        }
        out[(size_t)inst * HWSZ + row * WW + tid] = acc + s_b2;
    }
}

extern "C" void kernel_launch(void* const* d_in, const int* in_sizes, int n_in,
                              void* d_out, int out_size, void* d_ws, size_t ws_size,
                              hipStream_t stream) {
    (void)in_sizes; (void)n_in; (void)d_ws; (void)ws_size; (void)out_size;
    const float* x      = (const float*)d_in[0];   // [4, 64, 160, 256]
    const float* params = (const float*)d_in[1];   // [32, 8513]
    // d_in[2] = num_ins (static: all 8) — baked into the grid
    float* out = (float*)d_out;                    // [1, 32, 160, 256]

    dim3 grid(HH, 32);   // (row tile, instance)
    dim3 block(256);     // 8 waves of 32
    condlane_wmma_kernel<<<grid, block, 0, stream>>>(x, params, out);
}
